// ExpanderGatedGCNNet_81149112091153
// MI455X (gfx1250) — compile-verified, matrix operand order checked by reading, not run on hardware
//
#include <hip/hip_runtime.h>

// ---------------------------------------------------------------------------
// CDNA5 / gfx1250 GatedGCN forward.  wave32, WMMA bf16 16x16x32, f32 accum.
// LDS tiles padded to 132 ushorts/row (stride 66 words mod 64 banks = 2)
// to avoid 16-way bank conflicts on fragment ds_load_b128s.
// ---------------------------------------------------------------------------

typedef __attribute__((ext_vector_type(16))) __bf16 v16bf;
typedef __attribute__((ext_vector_type(8)))  float  v8f;

union FragAB { v16bf v; unsigned int u[8]; };
union FragC  { v8f   v; float        f[8]; };

constexpr int ROWP = 132;                      // padded row stride (ushorts)

__device__ __forceinline__ unsigned short f2bf(float f) {
    unsigned int u = __float_as_uint(f);
    u += 0x7FFFu + ((u >> 16) & 1u);           // round-to-nearest-even
    return (unsigned short)(u >> 16);
}
__device__ __forceinline__ unsigned int pack2bf(float lo, float hi) {
    return (unsigned int)f2bf(lo) | ((unsigned int)f2bf(hi) << 16);
}

// K offset within a 32-wide K chunk for A/B fragment register pair v (0..7).
__device__ __forceinline__ int kpair(int v, int hlf) {
    return (v < 4 ? 2 * v : 16 + 2 * (v - 4)) + hlf * 8;
}

// Stage a 128x128 fp32 tile as packed bf16 into padded LDS. 256 threads.
__device__ __forceinline__ void stageX(const float* __restrict__ X,
                                       unsigned short* sX, int blockBase, int M, int tid) {
    unsigned int* sXu = (unsigned int*)sX;
    for (int i = 0; i < 32; ++i) {             // 8192 uint pairs / 256 threads
        int idx = tid + i * 256;
        int row = idx >> 6, cp = idx & 63;
        int gm = blockBase + row;
        unsigned int pk = 0;
        if (gm < M) {
            const float* xp = &X[(size_t)gm * 128 + cp * 2];
            pk = pack2bf(xp[0], xp[1]);
        }
        sXu[row * (ROWP / 2) + cp] = pk;
    }
}
// Stage a 128x128 bf16 weight (already [n][k]) into padded LDS.
__device__ __forceinline__ void stageW(const unsigned short* __restrict__ Wt,
                                       unsigned short* sW, int tid) {
    const unsigned int* Wu = (const unsigned int*)Wt;
    unsigned int* sWu = (unsigned int*)sW;
    for (int i = 0; i < 32; ++i) {
        int idx = tid + i * 256;
        int row = idx >> 6, cp = idx & 63;
        sWu[row * (ROWP / 2) + cp] = Wu[row * 64 + cp];
    }
}

__device__ __forceinline__ void buildAfrags(const unsigned short* sX, FragAB* afrag,
                                            int rowbase, int mloc, int hlf) {
    const unsigned short* xr = &sX[(rowbase + mloc) * ROWP];
#pragma unroll
    for (int kc = 0; kc < 4; ++kc)
#pragma unroll
        for (int v = 0; v < 8; ++v)
            afrag[kc].u[v] = *(const unsigned int*)&xr[kc * 32 + kpair(v, hlf)];
}

__device__ __forceinline__ void tileGemm(const unsigned short* sW, const FragAB* afrag,
                                         FragC& acc, int n, int hlf) {
#pragma unroll
    for (int r = 0; r < 8; ++r) acc.f[r] = 0.f;
    const unsigned short* wr = &sW[n * ROWP];
#pragma unroll
    for (int kc = 0; kc < 4; ++kc) {
        FragAB bfrag;
#pragma unroll
        for (int v = 0; v < 8; ++v)
            bfrag.u[v] = *(const unsigned int*)&wr[kc * 32 + kpair(v, hlf)];
        acc.v = __builtin_amdgcn_wmma_f32_16x16x32_bf16(
            false, afrag[kc].v, false, bfrag.v, (short)0, acc.v, false, false);
    }
}

// ---------------------------------------------------------------------------
// Weight prep: fp32 W[k][n] (128x128) -> bf16 transposed Wt[n*128+k]
// ---------------------------------------------------------------------------
__global__ void convert_w(const float* __restrict__ W, unsigned short* __restrict__ Wt) {
    int idx = blockIdx.x * blockDim.x + threadIdx.x;
    if (idx >= 128 * 128) return;
    int n = idx >> 7, k = idx & 127;
    Wt[idx] = f2bf(W[k * 128 + n]);
}

__global__ void zero_f32(float* p, long n) {
    long i = (long)blockIdx.x * blockDim.x + threadIdx.x;
    long stride = (long)gridDim.x * blockDim.x;
    for (; i < n; i += stride) p[i] = 0.f;
}

// ---------------------------------------------------------------------------
// Y[M,128] = X[M,128] @ W + b   (single-weight version, for embeddings)
// ---------------------------------------------------------------------------
__global__ __launch_bounds__(256) void gemm128_bias(
    const float* __restrict__ X, const unsigned short* __restrict__ Wt,
    const float* __restrict__ bias, float* __restrict__ Y, int M)
{
    __shared__ unsigned short sX[128 * ROWP];
    __shared__ unsigned short sW[128 * ROWP];
    const int tid = threadIdx.x;
    const int blockBase = blockIdx.x * 128;

    stageX(X, sX, blockBase, M, tid);
    stageW(Wt, sW, tid);
    __syncthreads();

    const int wave = tid >> 5, lane = tid & 31;
    const int hlf = lane >> 4, mloc = lane & 15;
    const int rowbase = wave * 16;
    const bool full = (blockBase + 128) <= M;

    FragAB afrag[4];
    buildAfrags(sX, afrag, rowbase, mloc, hlf);

#pragma unroll
    for (int nt = 0; nt < 8; ++nt) {
        const int n = nt * 16 + mloc;
        FragC acc;
        tileGemm(sW, afrag, acc, n, hlf);
        const float b = bias[n];
        float* yp = &Y[(size_t)(blockBase + rowbase + 8 * hlf) * 128 + n];
        if (full) {
#pragma unroll
            for (int r = 0; r < 8; ++r) yp[(size_t)r * 128] = acc.f[r] + b;
        } else {
#pragma unroll
            for (int r = 0; r < 8; ++r) {
                int gm = blockBase + rowbase + r + 8 * hlf;
                if (gm < M) Y[(size_t)gm * 128 + n] = acc.f[r] + b;
            }
        }
    }
}

// ---------------------------------------------------------------------------
// Fused 4-weight node GEMM: stages X once, computes Y_j = X@W_j + b_j, j=0..3
// ---------------------------------------------------------------------------
__global__ __launch_bounds__(256) void gemm128_bias4(
    const float* __restrict__ X,
    const unsigned short* __restrict__ W0, const unsigned short* __restrict__ W1,
    const unsigned short* __restrict__ W2, const unsigned short* __restrict__ W3,
    const float* __restrict__ b0, const float* __restrict__ b1,
    const float* __restrict__ b2, const float* __restrict__ b3,
    float* __restrict__ Y0, float* __restrict__ Y1,
    float* __restrict__ Y2, float* __restrict__ Y3, int M)
{
    __shared__ unsigned short sX[128 * ROWP];
    __shared__ unsigned short sW[128 * ROWP];
    const int tid = threadIdx.x;
    const int blockBase = blockIdx.x * 128;

    const unsigned short* Ws[4] = {W0, W1, W2, W3};
    const float* bs[4] = {b0, b1, b2, b3};
    float* Ys[4] = {Y0, Y1, Y2, Y3};

    stageX(X, sX, blockBase, M, tid);

    const int wave = tid >> 5, lane = tid & 31;
    const int hlf = lane >> 4, mloc = lane & 15;
    const int rowbase = wave * 16;
    const bool full = (blockBase + 128) <= M;

    FragAB afrag[4];
    bool haveA = false;

    for (int j = 0; j < 4; ++j) {
        __syncthreads();                       // protect sW (and sX on j==0)
        stageW(Ws[j], sW, tid);
        __syncthreads();
        if (!haveA) { buildAfrags(sX, afrag, rowbase, mloc, hlf); haveA = true; }
        const float* bias = bs[j];
        float* Y = Ys[j];
#pragma unroll
        for (int nt = 0; nt < 8; ++nt) {
            const int n = nt * 16 + mloc;
            FragC acc;
            tileGemm(sW, afrag, acc, n, hlf);
            const float b = bias[n];
            float* yp = &Y[(size_t)(blockBase + rowbase + 8 * hlf) * 128 + n];
            if (full) {
#pragma unroll
                for (int r = 0; r < 8; ++r) yp[(size_t)r * 128] = acc.f[r] + b;
            } else {
#pragma unroll
                for (int r = 0; r < 8; ++r) {
                    int gm = blockBase + rowbase + r + 8 * hlf;
                    if (gm < M) Y[(size_t)gm * 128 + n] = acc.f[r] + b;
                }
            }
        }
    }
}

// ---------------------------------------------------------------------------
// Fused edge kernel: Ce = e@WC+bC (WMMA); e_new = Ce + Dh[src] + Eh[dst];
// sig = sigmoid(e_new); atomic num/den scatter at dst; t_e = e_new*snorm_e;
// per-feature BN sum/sumsq accumulated (LDS then one atomic flush).
// ---------------------------------------------------------------------------
__global__ __launch_bounds__(256) void edge_fused(
    const float* __restrict__ e, const unsigned short* __restrict__ WtC,
    const float* __restrict__ bC,
    const float* __restrict__ Bmat, const float* __restrict__ Dmat,
    const float* __restrict__ Emat,
    const int* __restrict__ src, const int* __restrict__ dst,
    const float* __restrict__ snorm_e,
    float* __restrict__ t_e, float* __restrict__ num, float* __restrict__ den,
    float* __restrict__ stats /* [0:128) sum, [128:256) sumsq */, int Ecount)
{
    __shared__ unsigned short sX[128 * ROWP];
    __shared__ unsigned short sW[128 * ROWP];
    __shared__ float sSum[128], sSq[128];
    const int tid = threadIdx.x;
    const int blockBase = blockIdx.x * 128;

    stageX(e, sX, blockBase, Ecount, tid);
    stageW(WtC, sW, tid);
    if (tid < 128) { sSum[tid] = 0.f; sSq[tid] = 0.f; }
    __syncthreads();

    const int wave = tid >> 5, lane = tid & 31;
    const int hlf = lane >> 4, mloc = lane & 15;
    const int rowbase = wave * 16;
    const bool full = (blockBase + 128) <= Ecount;

    FragAB afrag[4];
    buildAfrags(sX, afrag, rowbase, mloc, hlf);

    int   srcs[8], dsts[8];
    float sn[8];
#pragma unroll
    for (int r = 0; r < 8; ++r) {
        int eI = blockBase + rowbase + r + 8 * hlf;
        bool ok = full || (eI < Ecount);
        srcs[r] = ok ? src[eI] : 0;
        dsts[r] = ok ? dst[eI] : 0;
        sn[r]   = ok ? snorm_e[eI] : 0.f;
    }

#pragma unroll
    for (int nt = 0; nt < 8; ++nt) {
        const int n = nt * 16 + mloc;
        FragC acc;
        tileGemm(sW, afrag, acc, n, hlf);
        const float bc = bC[n];
        float lsum = 0.f, lsq = 0.f;
        if (full) {
#pragma unroll
            for (int r = 0; r < 8; ++r) {
                int eI = blockBase + rowbase + r + 8 * hlf;
                float enew = acc.f[r] + bc
                           + Dmat[(size_t)srcs[r] * 128 + n]
                           + Emat[(size_t)dsts[r] * 128 + n];
                float sg = 1.f / (1.f + __expf(-enew));
                atomicAdd(&num[(size_t)dsts[r] * 128 + n],
                          sg * Bmat[(size_t)srcs[r] * 128 + n]);
                atomicAdd(&den[(size_t)dsts[r] * 128 + n], sg);
                float t = enew * sn[r];
                t_e[(size_t)eI * 128 + n] = t;
                lsum += t; lsq += t * t;
            }
        } else {
#pragma unroll
            for (int r = 0; r < 8; ++r) {
                int eI = blockBase + rowbase + r + 8 * hlf;
                if (eI < Ecount) {
                    float enew = acc.f[r] + bc
                               + Dmat[(size_t)srcs[r] * 128 + n]
                               + Emat[(size_t)dsts[r] * 128 + n];
                    float sg = 1.f / (1.f + __expf(-enew));
                    atomicAdd(&num[(size_t)dsts[r] * 128 + n],
                              sg * Bmat[(size_t)srcs[r] * 128 + n]);
                    atomicAdd(&den[(size_t)dsts[r] * 128 + n], sg);
                    float t = enew * sn[r];
                    t_e[(size_t)eI * 128 + n] = t;
                    lsum += t; lsq += t * t;
                }
            }
        }
        atomicAdd(&sSum[n], lsum);
        atomicAdd(&sSq[n], lsq);
    }
    __syncthreads();
    if (tid < 128) {
        atomicAdd(&stats[tid],       sSum[tid]);
        atomicAdd(&stats[128 + tid], sSq[tid]);
    }
}

// h_new = Ah + num/(den+1e-6); t_h = h_new*snorm_n; BN stats accumulated.
__global__ __launch_bounds__(128) void node_msg(
    const float* __restrict__ Ah, const float* __restrict__ num,
    const float* __restrict__ den, const float* __restrict__ snorm_n,
    float* __restrict__ t_h, float* __restrict__ stats, int Nn)
{
    const int n = threadIdx.x;
    float s = 0.f, q = 0.f;
    for (int row = blockIdx.x; row < Nn; row += gridDim.x) {
        size_t idx = (size_t)row * 128 + n;
        float v = Ah[idx] + num[idx] / (den[idx] + 1e-6f);
        float t = v * snorm_n[row];
        t_h[idx] = t;
        s += t; q += t * t;
    }
    atomicAdd(&stats[n], s);
    atomicAdd(&stats[128 + n], q);
}

// base += relu( (t - mu)*rsqrt(var+1e-5)*gamma + beta )
__global__ void bn_residual(float* __restrict__ base, const float* __restrict__ t,
                            const float* __restrict__ stats,
                            const float* __restrict__ gamma, const float* __restrict__ beta,
                            float invM, long total)
{
    long i = (long)blockIdx.x * blockDim.x + threadIdx.x;
    long stride = (long)gridDim.x * blockDim.x;
    for (; i < total; i += stride) {
        int n = (int)(i & 127);
        float mu  = stats[n] * invM;
        float var = stats[128 + n] * invM - mu * mu;
        float y = (t[i] - mu) * rsqrtf(var + 1e-5f) * gamma[n] + beta[n];
        base[i] += fmaxf(y, 0.f);
    }
}

__global__ __launch_bounds__(128) void readout_accum(
    const float* __restrict__ h, const int* __restrict__ gid,
    float* __restrict__ gsum, float* __restrict__ gcnt, int Nn)
{
    const int n = threadIdx.x;
    for (int row = blockIdx.x; row < Nn; row += gridDim.x) {
        int g = gid[row];
        atomicAdd(&gsum[(size_t)g * 128 + n], h[(size_t)row * 128 + n]);
        if (n == 0) atomicAdd(&gcnt[g], 1.f);
    }
}

__global__ void readout_final(const float* __restrict__ gsum, const float* __restrict__ gcnt,
                              const float* __restrict__ Wr, const float* __restrict__ br,
                              float* __restrict__ out, int Gn, int Cn)
{
    int tid = blockIdx.x * blockDim.x + threadIdx.x;
    if (tid >= Gn * Cn) return;
    int g = tid / Cn, c = tid % Cn;
    float cnt = fmaxf(gcnt[g], 1.f);
    float s = 0.f;
    for (int k = 0; k < 128; ++k) s += (gsum[g * 128 + k] / cnt) * Wr[k * Cn + c];
    out[tid] = s + br[c];
}

// ---------------------------------------------------------------------------
extern "C" void kernel_launch(void* const* d_in, const int* in_sizes, int n_in,
                              void* d_out, int out_size, void* d_ws, size_t ws_size,
                              hipStream_t stream)
{
    const float* h_in    = (const float*)d_in[0];
    const float* e_in    = (const float*)d_in[1];
    const float* snorm_n = (const float*)d_in[2];
    const float* snorm_e = (const float*)d_in[3];
    const int*   src     = (const int*)d_in[4];
    const int*   dst     = (const int*)d_in[5];
    const int*   gid     = (const int*)d_in[6];
    const float* Wh_emb  = (const float*)d_in[7];
    const float* bh_emb  = (const float*)d_in[8];
    const float* We_emb  = (const float*)d_in[9];
    const float* be_emb  = (const float*)d_in[10];
    const float* WA = (const float*)d_in[11]; const float* bA = (const float*)d_in[12];
    const float* WB = (const float*)d_in[13]; const float* bB = (const float*)d_in[14];
    const float* WC = (const float*)d_in[15]; const float* bC = (const float*)d_in[16];
    const float* WD = (const float*)d_in[17]; const float* bD = (const float*)d_in[18];
    const float* WE = (const float*)d_in[19]; const float* bE = (const float*)d_in[20];
    const float* gn_h = (const float*)d_in[21]; const float* bt_h = (const float*)d_in[22];
    const float* gn_e = (const float*)d_in[23]; const float* bt_e = (const float*)d_in[24];
    const float* Wr = (const float*)d_in[25]; const float* br = (const float*)d_in[26];

    const int Nn = in_sizes[0] / 128;
    const int Ee = in_sizes[1] / 128;
    const int Ll = in_sizes[11] / (128 * 128);
    const int Cc = in_sizes[26];
    const int Gg = out_size / Cc;

    // ---- workspace carve-out ----
    char* wsBase = (char*)d_ws;
    size_t off = 0;
    auto take = [&](size_t bytes) -> void* {
        void* p = wsBase + off;
        off += (bytes + 255) & ~(size_t)255;
        return p;
    };
    const int NW = 2 + 5 * Ll;
    unsigned short* wbf = (unsigned short*)take((size_t)NW * 16384 * sizeof(unsigned short));
    float* hbuf = (float*)take((size_t)Nn * 128 * 4);
    float* ebuf = (float*)take((size_t)Ee * 128 * 4);
    float* Ahb  = (float*)take((size_t)Nn * 128 * 4);
    float* Bhb  = (float*)take((size_t)Nn * 128 * 4);
    float* Dhb  = (float*)take((size_t)Nn * 128 * 4);
    float* Ehb  = (float*)take((size_t)Nn * 128 * 4);
    float* thb  = (float*)take((size_t)Nn * 128 * 4);
    float* numb = (float*)take((size_t)Nn * 128 * 4);
    float* denb = (float*)take((size_t)Nn * 128 * 4);
    float* teb  = (float*)take((size_t)Ee * 128 * 4);
    float* stats = (float*)take(512 * 4);      // [0:256) h stats, [256:512) e stats
    float* gsum  = (float*)take((size_t)Gg * 128 * 4);
    float* gcnt  = (float*)take((size_t)Gg * 4);
    (void)ws_size; (void)n_in;

    const int wgrid = (16384 + 255) / 256;
    auto wslot = [&](int s) { return wbf + (size_t)s * 16384; };

    // ---- weight conversion (fp32 -> bf16, transposed) ----
    convert_w<<<wgrid, 256, 0, stream>>>(Wh_emb, wslot(0));
    convert_w<<<wgrid, 256, 0, stream>>>(We_emb, wslot(1));
    for (int l = 0; l < Ll; ++l) {
        convert_w<<<wgrid, 256, 0, stream>>>(WA + (size_t)l * 16384, wslot(2 + 5 * l + 0));
        convert_w<<<wgrid, 256, 0, stream>>>(WB + (size_t)l * 16384, wslot(2 + 5 * l + 1));
        convert_w<<<wgrid, 256, 0, stream>>>(WC + (size_t)l * 16384, wslot(2 + 5 * l + 2));
        convert_w<<<wgrid, 256, 0, stream>>>(WD + (size_t)l * 16384, wslot(2 + 5 * l + 3));
        convert_w<<<wgrid, 256, 0, stream>>>(WE + (size_t)l * 16384, wslot(2 + 5 * l + 4));
    }

    const int ngrid = (Nn + 127) / 128;
    const int egrid = (Ee + 127) / 128;

    // ---- input embeddings ----
    gemm128_bias<<<ngrid, 256, 0, stream>>>(h_in, wslot(0), bh_emb, hbuf, Nn);
    gemm128_bias<<<egrid, 256, 0, stream>>>(e_in, wslot(1), be_emb, ebuf, Ee);

    long nTot = (long)Nn * 128, eTot = (long)Ee * 128;
    int nBlocks = (int)((nTot + 255) / 256); if (nBlocks > 16384) nBlocks = 16384;
    int eBlocks = (int)((eTot + 255) / 256); if (eBlocks > 65536) eBlocks = 65536;

    // ---- layers ----
    for (int l = 0; l < Ll; ++l) {
        zero_f32<<<nBlocks, 256, 0, stream>>>(numb, nTot);
        zero_f32<<<nBlocks, 256, 0, stream>>>(denb, nTot);
        zero_f32<<<1, 256, 0, stream>>>(stats, 512);

        gemm128_bias4<<<ngrid, 256, 0, stream>>>(
            hbuf,
            wslot(2 + 5 * l + 0), wslot(2 + 5 * l + 1),
            wslot(2 + 5 * l + 3), wslot(2 + 5 * l + 4),
            bA + l * 128, bB + l * 128, bD + l * 128, bE + l * 128,
            Ahb, Bhb, Dhb, Ehb, Nn);

        edge_fused<<<egrid, 256, 0, stream>>>(ebuf, wslot(2 + 5 * l + 2), bC + l * 128,
                                              Bhb, Dhb, Ehb, src, dst, snorm_e,
                                              teb, numb, denb, stats + 256, Ee);

        node_msg<<<480, 128, 0, stream>>>(Ahb, numb, denb, snorm_n, thb, stats, Nn);

        bn_residual<<<nBlocks, 256, 0, stream>>>(hbuf, thb, stats,
                                                 gn_h + l * 128, bt_h + l * 128,
                                                 1.0f / (float)Nn, nTot);
        bn_residual<<<eBlocks, 256, 0, stream>>>(ebuf, teb, stats + 256,
                                                 gn_e + l * 128, bt_e + l * 128,
                                                 1.0f / (float)Ee, eTot);
    }

    // ---- readout ----
    zero_f32<<<(Gg * 128 + 255) / 256, 256, 0, stream>>>(gsum, (long)Gg * 128);
    zero_f32<<<1, 256, 0, stream>>>(gcnt, Gg);
    readout_accum<<<480, 128, 0, stream>>>(hbuf, gid, gsum, gcnt, Nn);
    readout_final<<<(Gg * Cc + 255) / 256, 256, 0, stream>>>(gsum, gcnt, Wr, br,
                                                            (float*)d_out, Gg, Cc);
}
